// MoEGPT_46291157516848
// MI455X (gfx1250) — compile-verified, matrix operand order checked by reading, not run on hardware
//
#include <hip/hip_runtime.h>
#include <hip/hip_bf16.h>
#include <math.h>

// ---------------------------------------------------------------------------
// MoE-GPT forward for MI455X (gfx1250).  All heavy GEMMs run on the WMMA path
// (v_wmma_f32_16x16x32_f16) with f32->f16 conversion at LDS-fill time, a
// one-tile register prefetch pipeline, and double-buffered LDS so each K-step
// needs only a single workgroup barrier.
// Shapes: B=8 T=128 (N=1024 tokens), C=768, L=12, H=12, D=64, E=8, F=3072,
// V=50257.
// ---------------------------------------------------------------------------

typedef _Float16 halfx4  __attribute__((ext_vector_type(4)));
typedef _Float16 halfx8  __attribute__((ext_vector_type(8)));
typedef _Float16 v16h    __attribute__((ext_vector_type(16)));
typedef float    v8f     __attribute__((ext_vector_type(8)));

#define NTOK 1024
#define CDIM 768
#define FDIM 3072
#define EEXP 8
#define LDT  40           // LDS row pitch in halves (80B: 16B aligned, spreads banks)
#define BUFH (128 * LDT)  // halves per (A or B) buffer

// Load one 16x32 f16 fragment (A or B operand) from an LDS tile.
// Layout per CDNA5 ISA 7.12.2: lanes 0-15 -> rows 0-15 with K in two 8-half
// runs {0..7, 16..23}; lanes 16-31 -> same rows with K runs {8..15, 24..31}.
__device__ __forceinline__ v16h frag_ld(const _Float16* rowbase, int lane) {
  const _Float16* p = rowbase + (lane & 15) * LDT + ((lane >> 4) << 3);
  halfx8 lo = *(const halfx8*)(p);
  halfx8 hi = *(const halfx8*)(p + 16);
  v16h r;
#pragma unroll
  for (int i = 0; i < 8; ++i) { r[i] = lo[i]; r[i + 8] = hi[i]; }
  return r;
}

__device__ __forceinline__ halfx4 cvt4(float4 v) {
  halfx4 h = { (_Float16)v.x, (_Float16)v.y, (_Float16)v.z, (_Float16)v.w };
  return h;
}

// C[M,N] = (acc? C : 0) + A[M,K] * B[N,K]^T   (all f32 row-major in HBM)
// Exploited: M % 128 == 0 (always 1024 here), K % 32 == 0.  Only N may be
// ragged (logits, N=50257): handled branch-free by clamping B row indices on
// load and guarding columns on store.
// Block = 128x128 tile, 256 threads = 8 wave32 in a 4(M) x 2(N) grid,
// each wave owns 32x64 = eight 16x16 WMMA accumulators.
__global__ __launch_bounds__(256) void gemm_wmma_k(
    const float* __restrict__ A, const float* __restrict__ Bw,
    float* __restrict__ C, int M, int N, int K,
    int lda, int ldb, int ldc, int accumulate)
{
  __shared__ __align__(16) _Float16 As[2][128][LDT];
  __shared__ __align__(16) _Float16 Bs[2][128][LDT];

  const int tid  = threadIdx.x;
  const int lane = tid & 31;
  const int wave = tid >> 5;
  const int wm   = (wave & 3) * 32;   // wave row offset in block tile
  const int wn   = (wave >> 2) * 64;  // wave col offset in block tile
  const long bm  = (long)blockIdx.x * 128;
  const long bn  = (long)blockIdx.y * 128;

  // ---- per-thread staging addresses (computed once) --------------------
  // Each thread moves 4 float4 of A and 4 of B per K-step (128x32 panels).
  int   oA[4], oB[4];                 // 32-bit element offsets from A / Bw
  _Float16* sA[4];                    // LDS targets in buffer 0
  _Float16* sB[4];
#pragma unroll
  for (int u = 0; u < 4; ++u) {
    int id = tid + u * 256;           // 0..1023
    int r  = id >> 3;                 // 0..127
    int c4 = (id & 7) << 2;           // 0,4,...,28
    long ga = bm + r;                 // M % 128 == 0 -> always in range
    long gb = bn + r;
    if (gb > (long)N - 1) gb = (long)N - 1;   // clamp (branch-free edge)
    oA[u] = (int)(ga * lda + c4);
    oB[u] = (int)(gb * ldb + c4);
    sA[u] = &As[0][r][c4];
    sB[u] = &Bs[0][r][c4];
  }

  const v8f vzero = {0.f, 0.f, 0.f, 0.f, 0.f, 0.f, 0.f, 0.f};
  v8f acc[2][4];
#pragma unroll
  for (int i = 0; i < 2; ++i)
#pragma unroll
    for (int j = 0; j < 4; ++j) acc[i][j] = vzero;

  const int nk = K >> 5;              // number of 32-wide K tiles (K%32==0)
  float4 pa[4], pb[4];

  // ---- prologue: tile 0 -> regs -> buf0; tile 1 -> regs ----------------
#pragma unroll
  for (int u = 0; u < 4; ++u) {
    pa[u] = *(const float4*)(A  + oA[u]);
    pb[u] = *(const float4*)(Bw + oB[u]);
    oA[u] += 32;
    oB[u] += 32;
  }
#pragma unroll
  for (int u = 0; u < 4; ++u) {
    *(halfx4*)sA[u] = cvt4(pa[u]);
    *(halfx4*)sB[u] = cvt4(pb[u]);
  }
  if (nk > 1) {
#pragma unroll
    for (int u = 0; u < 4; ++u) {
      pa[u] = *(const float4*)(A  + oA[u]);
      pb[u] = *(const float4*)(Bw + oB[u]);
      oA[u] += 32;
      oB[u] += 32;
    }
  }
  __syncthreads();

  for (int t = 0; t < nk; ++t) {
    const int cur = t & 1;
    const int nxt = cur ^ 1;

    // ---- commit regs (tile t+1) to the other buffer, start tile t+2 ----
    if (t + 1 < nk) {
      const int boff = nxt * BUFH;
#pragma unroll
      for (int u = 0; u < 4; ++u) {
        *(halfx4*)(sA[u] + boff) = cvt4(pa[u]);   // waits loadcnt; covered
        *(halfx4*)(sB[u] + boff) = cvt4(pb[u]);   // by tile t-1's compute
      }
      if (t + 2 < nk) {
#pragma unroll
        for (int u = 0; u < 4; ++u) {
          pa[u] = *(const float4*)(A  + oA[u]);
          pb[u] = *(const float4*)(Bw + oB[u]);
          oA[u] += 32;
          oB[u] += 32;
        }
      }
    }

    // ---- compute on current buffer ----
    v16h a0 = frag_ld(&As[cur][wm +  0][0], lane);
    v16h a1 = frag_ld(&As[cur][wm + 16][0], lane);
    v16h b0 = frag_ld(&Bs[cur][wn +  0][0], lane);
    v16h b1 = frag_ld(&Bs[cur][wn + 16][0], lane);
    v16h b2 = frag_ld(&Bs[cur][wn + 32][0], lane);
    v16h b3 = frag_ld(&Bs[cur][wn + 48][0], lane);

    acc[0][0] = __builtin_amdgcn_wmma_f32_16x16x32_f16(false, a0, false, b0, (short)0, acc[0][0], false, false);
    acc[0][1] = __builtin_amdgcn_wmma_f32_16x16x32_f16(false, a0, false, b1, (short)0, acc[0][1], false, false);
    acc[0][2] = __builtin_amdgcn_wmma_f32_16x16x32_f16(false, a0, false, b2, (short)0, acc[0][2], false, false);
    acc[0][3] = __builtin_amdgcn_wmma_f32_16x16x32_f16(false, a0, false, b3, (short)0, acc[0][3], false, false);
    acc[1][0] = __builtin_amdgcn_wmma_f32_16x16x32_f16(false, a1, false, b0, (short)0, acc[1][0], false, false);
    acc[1][1] = __builtin_amdgcn_wmma_f32_16x16x32_f16(false, a1, false, b1, (short)0, acc[1][1], false, false);
    acc[1][2] = __builtin_amdgcn_wmma_f32_16x16x32_f16(false, a1, false, b2, (short)0, acc[1][2], false, false);
    acc[1][3] = __builtin_amdgcn_wmma_f32_16x16x32_f16(false, a1, false, b3, (short)0, acc[1][3], false, false);

    __syncthreads();   // single barrier per K-step (double-buffered LDS)
  }

  // ---- store: C/D layout (7.12.2): vgpr v -> row v + 8*(lane/16), col lane%16
  const int rowsel = (lane >> 4) * 8;
  const int coloff = lane & 15;
#pragma unroll
  for (int i = 0; i < 2; ++i) {
#pragma unroll
    for (int j = 0; j < 4; ++j) {
      long col = bn + wn + j * 16 + coloff;
      if (col >= N) continue;
      long rb = bm + wm + i * 16 + rowsel;
#pragma unroll
      for (int v = 0; v < 8; ++v) {
        long row = rb + v;
        float* p = C + row * (long)ldc + col;
        float val = acc[i][j][v];
        *p = accumulate ? (*p + val) : val;
      }
    }
  }
}

// x[n,c] = tok_emb[idx[n],c] + pos_emb[n%T,c]
__global__ __launch_bounds__(256) void embed_k(const int* __restrict__ idx,
                                               const float* __restrict__ te,
                                               const float* __restrict__ pe,
                                               float* __restrict__ x) {
  int i = blockIdx.x * 256 + threadIdx.x;   // < 1024*768
  int n = i / CDIM, c = i % CDIM;
  x[i] = te[(long)idx[n] * CDIM + c] + pe[(n & 127) * CDIM + c];
}

// One block per row; 256 threads * 3 elements = C=768.
__global__ __launch_bounds__(256) void ln_k(const float* __restrict__ x,
                                            const float* __restrict__ g,
                                            const float* __restrict__ b,
                                            float* __restrict__ y) {
  int n = blockIdx.x, tid = threadIdx.x;
  const float* xr = x + (long)n * CDIM;
  float v0 = xr[tid], v1 = xr[tid + 256], v2 = xr[tid + 512];
  __shared__ float red[256];
  red[tid] = v0 + v1 + v2;
  __syncthreads();
  for (int st = 128; st > 0; st >>= 1) {
    if (tid < st) red[tid] += red[tid + st];
    __syncthreads();
  }
  float mu = red[0] * (1.f / 768.f);
  __syncthreads();
  float d0 = v0 - mu, d1 = v1 - mu, d2 = v2 - mu;
  red[tid] = d0 * d0 + d1 * d1 + d2 * d2;
  __syncthreads();
  for (int st = 128; st > 0; st >>= 1) {
    if (tid < st) red[tid] += red[tid + st];
    __syncthreads();
  }
  float rstd = rsqrtf(red[0] * (1.f / 768.f) + 1e-5f);
  float* yr = y + (long)n * CDIM;
  yr[tid]       = d0 * rstd * g[tid]       + b[tid];
  yr[tid + 256] = d1 * rstd * g[tid + 256] + b[tid + 256];
  yr[tid + 512] = d2 * rstd * g[tid + 512] + b[tid + 512];
}

// Causal attention, one block per (b,h), thread t = query row, online softmax.
// K/V rows staged in LDS with a +row swizzle to spread banks.
__global__ __launch_bounds__(128) void attn_k(const float* __restrict__ qkv,
                                              float* __restrict__ out) {
  int b = blockIdx.x, hh = blockIdx.y, t = threadIdx.x;
  __shared__ float Ks[128][64];
  __shared__ float Vs[128][64];
  const float* base = qkv + (long)(b * 128 + t) * (3 * CDIM);
#pragma unroll 8
  for (int d = 0; d < 64; ++d) {
    Ks[t][(d + t) & 63] = base[CDIM     + hh * 64 + d];
    Vs[t][(d + t) & 63] = base[2 * CDIM + hh * 64 + d];
  }
  float q[64];
#pragma unroll 8
  for (int d = 0; d < 64; ++d) q[d] = base[hh * 64 + d] * 0.125f;  // D^-0.5
  __syncthreads();

  float m = -3.0e38f, s = 0.f;
  float acc[64];
#pragma unroll 8
  for (int d = 0; d < 64; ++d) acc[d] = 0.f;
  for (int j = 0; j <= t; ++j) {
    float dot = 0.f;
#pragma unroll 8
    for (int d = 0; d < 64; ++d) dot += q[d] * Ks[j][(d + j) & 63];
    float mn   = fmaxf(m, dot);
    float corr = __expf(m - mn);
    float p    = __expf(dot - mn);
    s = s * corr + p;
#pragma unroll 8
    for (int d = 0; d < 64; ++d) acc[d] = acc[d] * corr + p * Vs[j][(d + j) & 63];
    m = mn;
  }
  float inv = 1.f / s;
  float* o = out + (long)(b * 128 + t) * CDIM + hh * 64;
#pragma unroll 8
  for (int d = 0; d < 64; ++d) o[d] = acc[d] * inv;
}

// One block per token: router logits (E=8 dots of length 768), softmax,
// top-2 renormalized combine weights scattered into comb[n,0..7].
__global__ __launch_bounds__(128) void router_k(const float* __restrict__ xn,
                                                const float* __restrict__ rw,
                                                float* __restrict__ comb) {
  int n = blockIdx.x, tid = threadIdx.x;
  __shared__ float red[128];
  __shared__ float logit[EEXP];
  const float* xr = xn + (long)n * CDIM;
  for (int e = 0; e < EEXP; ++e) {
    float p = 0.f;
    for (int c = tid; c < CDIM; c += 128) p += xr[c] * rw[e * CDIM + c];
    red[tid] = p;
    __syncthreads();
    for (int st = 64; st > 0; st >>= 1) {
      if (tid < st) red[tid] += red[tid + st];
      __syncthreads();
    }
    if (tid == 0) logit[e] = red[0];
    __syncthreads();
  }
  if (tid == 0) {
    float mx = logit[0];
    for (int e = 1; e < EEXP; ++e) mx = fmaxf(mx, logit[e]);
    float p[EEXP], s = 0.f;
    for (int e = 0; e < EEXP; ++e) { p[e] = __expf(logit[e] - mx); s += p[e]; }
    float inv = 1.f / s;
    for (int e = 0; e < EEXP; ++e) p[e] *= inv;
    int i1 = 0;
    for (int e = 1; e < EEXP; ++e) if (p[e] > p[i1]) i1 = e;
    int i2 = (i1 == 0) ? 1 : 0;
    for (int e = 0; e < EEXP; ++e) if (e != i1 && p[e] > p[i2]) i2 = e;
    float t2 = p[i1] + p[i2];
    for (int e = 0; e < EEXP; ++e) comb[n * EEXP + e] = 0.f;
    comb[n * EEXP + i1] = p[i1] / t2;
    comb[n * EEXP + i2] = p[i2] / t2;
  }
}

// x[n,c] += sum_e comb[n,e] * b2[e,c]   (folds the w2 bias under combine)
__global__ __launch_bounds__(256) void biascomb_k(float* __restrict__ x,
                                                  const float* __restrict__ comb,
                                                  const float* __restrict__ b2l) {
  int i = blockIdx.x * 256 + threadIdx.x;   // < 1024*768
  int n = i / CDIM, c = i % CDIM;
  float s = 0.f;
#pragma unroll
  for (int e = 0; e < EEXP; ++e) s += comb[n * EEXP + e] * b2l[e * CDIM + c];
  x[i] += s;
}

// h[n,f] = gelu(h[n,f] + b1[e,f]) * comb[n,e]  (exact gelu; combine folded in
// so the w2 GEMM can accumulate straight into the residual stream)
__global__ __launch_bounds__(256) void gelu_k(float* __restrict__ h,
                                              const float* __restrict__ b1e,
                                              const float* __restrict__ comb,
                                              int e) {
  long i = (long)blockIdx.x * 256 + threadIdx.x;  // < 1024*3072
  int n = (int)(i / FDIM), f = (int)(i % FDIM);
  float v = h[i] + b1e[f];
  v = 0.5f * v * (1.f + erff(v * 0.70710678118f));
  h[i] = v * comb[n * EEXP + e];
}

static void gemm(const float* A, const float* B, float* C,
                 int M, int N, int K, int accumulate, hipStream_t s) {
  dim3 grid((M + 127) / 128, (N + 127) / 128);
  gemm_wmma_k<<<grid, 256, 0, s>>>(A, B, C, M, N, K, K, K, N, accumulate);
}

extern "C" void kernel_launch(void* const* d_in, const int* in_sizes, int n_in,
                              void* d_out, int out_size, void* d_ws, size_t ws_size,
                              hipStream_t stream) {
  (void)in_sizes; (void)n_in; (void)out_size; (void)ws_size;
  const int*   idx      = (const int*)  d_in[0];
  const float* tok_emb  = (const float*)d_in[1];
  const float* pos_emb  = (const float*)d_in[2];
  const float* ln1_s    = (const float*)d_in[3];
  const float* ln1_b    = (const float*)d_in[4];
  const float* qkv_w    = (const float*)d_in[5];
  const float* proj_w   = (const float*)d_in[6];
  const float* ln2_s    = (const float*)d_in[7];
  const float* ln2_b    = (const float*)d_in[8];
  const float* router_w = (const float*)d_in[9];
  const float* w1       = (const float*)d_in[10];
  const float* b1       = (const float*)d_in[11];
  const float* w2       = (const float*)d_in[12];
  const float* b2       = (const float*)d_in[13];
  const float* lnf_s    = (const float*)d_in[14];
  const float* lnf_b    = (const float*)d_in[15];
  float* out = (float*)d_out;

  // Workspace layout (f32): ~31.5 MB total.
  float* x    = (float*)d_ws;
  float* xn   = x    + (long)NTOK * CDIM;
  float* qkv  = xn   + (long)NTOK * CDIM;
  float* attn = qkv  + (long)NTOK * 3 * CDIM;
  float* h    = attn + (long)NTOK * CDIM;
  float* comb = h    + (long)NTOK * FDIM;

  embed_k<<<NTOK * CDIM / 256, 256, 0, stream>>>(idx, tok_emb, pos_emb, x);

  for (int l = 0; l < 12; ++l) {
    ln_k<<<NTOK, 256, 0, stream>>>(x, ln1_s + l * CDIM, ln1_b + l * CDIM, xn);
    gemm(xn, qkv_w + (long)l * 3 * CDIM * CDIM, qkv, NTOK, 3 * CDIM, CDIM, 0, stream);
    attn_k<<<dim3(8, 12), 128, 0, stream>>>(qkv, attn);
    gemm(attn, proj_w + (long)l * CDIM * CDIM, x, NTOK, CDIM, CDIM, 1, stream);  // x += proj

    ln_k<<<NTOK, 256, 0, stream>>>(x, ln2_s + l * CDIM, ln2_b + l * CDIM, xn);
    router_k<<<NTOK, 128, 0, stream>>>(xn, router_w + (long)l * EEXP * CDIM, comb);
    biascomb_k<<<NTOK * CDIM / 256, 256, 0, stream>>>(x, comb, b2 + (long)l * EEXP * CDIM);

    for (int e = 0; e < EEXP; ++e) {
      long we = (long)l * EEXP + e;
      gemm(xn, w1 + we * (long)FDIM * CDIM, h, NTOK, FDIM, CDIM, 0, stream);
      gelu_k<<<NTOK * FDIM / 256, 256, 0, stream>>>(h, b1 + we * FDIM, comb, e);
      gemm(h, w2 + we * (long)CDIM * FDIM, x, NTOK, CDIM, FDIM, 1, stream);    // x += c*y_e
    }
  }

  ln_k<<<NTOK, 256, 0, stream>>>(x, lnf_s, lnf_b, xn);
  gemm(xn, tok_emb, out, NTOK, 50257, CDIM, 0, stream);  // tied head
}
